// MiniYOLAF_53249004536106
// MI455X (gfx1250) — matrix operand order repaired and checked by reading, not compiled, not run on hardware
//
#include <hip/hip_runtime.h>
#include <hip/hip_bf16.h>
#include <math.h>

typedef __attribute__((ext_vector_type(16))) _Float16 v16h;
typedef __attribute__((ext_vector_type(8)))  _Float16 v8h;
typedef __attribute__((ext_vector_type(8)))  float    v8f;

// ---------------------------------------------------------------------------
// Utility kernels
// ---------------------------------------------------------------------------

__global__ void k_zero32(uint32_t* __restrict__ p, size_t n32) {
    size_t i = (size_t)blockIdx.x * blockDim.x + threadIdx.x;
    size_t st = (size_t)gridDim.x * blockDim.x;
    for (; i < n32; i += st) p[i] = 0u;
}

// f32 NCHW (batch 0) -> f16 NHWC (unpadded)
__global__ void k_nchw_to_nhwc_f16(const float* __restrict__ in, _Float16* __restrict__ out,
                                   int C, int H, int W) {
    int n = C * H * W;
    int i = blockIdx.x * blockDim.x + threadIdx.x;
    if (i >= n) return;
    int c = i % C;
    int x = (i / C) % W;
    int y = i / (C * W);
    out[i] = (_Float16)in[((size_t)c * H + y) * W + x];   // out idx == (y*W+x)*C + c
}

// f32 OIHW weights -> f16 [OC16][tap][IC], zero-padded rows; bias -> f32 padded
__global__ void k_cvt_w(const float* __restrict__ w, const float* __restrict__ b,
                        _Float16* __restrict__ w16, float* __restrict__ b16,
                        int OC, int OC16, int IC, int K) {
    int taps = K * K;
    int n = OC16 * taps * IC;
    int i = blockIdx.x * blockDim.x + threadIdx.x;
    if (i < OC16) b16[i] = (i < OC) ? b[i] : 0.f;
    if (i >= n) return;
    int ic  = i % IC;
    int tap = (i / IC) % taps;
    int oc  = i / (IC * taps);
    float v = 0.f;
    if (oc < OC) v = w[(((size_t)oc * IC + ic) * K + tap / K) * K + (tap % K)];
    w16[i] = (_Float16)v;
}

// f32 NHWC (stride C) -> f16 NHWC padded(+1) interior
__global__ void k_pack_f16pad(const float* __restrict__ in, _Float16* __restrict__ out,
                              int C, int H, int W) {
    int n = C * H * W;
    int i = blockIdx.x * blockDim.x + threadIdx.x;
    if (i >= n) return;
    int c = i % C;
    int x = (i / C) % W;
    int y = i / (C * W);
    out[((size_t)(y + 1) * (W + 2) + (x + 1)) * C + c] = (_Float16)in[i];
}

// big += bilinear(sm)  (align-corners style, matches jnp.linspace sampling)
__global__ void k_upsample_add(const float* __restrict__ sm, float* __restrict__ big,
                               int C, int h, int w, int H, int W) {
    int n = C * H * W;
    int i = blockIdx.x * blockDim.x + threadIdx.x;
    if (i >= n) return;
    int c = i % C;
    int X = (i / C) % W;
    int Y = i / (C * W);
    float sy = (H > 1) ? (float)(h - 1) / (float)(H - 1) : 0.f;
    float sx = (W > 1) ? (float)(w - 1) / (float)(W - 1) : 0.f;
    float ys = Y * sy, xs = X * sx;
    int y0 = (int)floorf(ys); int y1 = (y0 + 1 < h - 1) ? y0 + 1 : h - 1; float fy = ys - y0;
    int x0 = (int)floorf(xs); int x1 = (x0 + 1 < w - 1) ? x0 + 1 : w - 1; float fx = xs - x0;
    float v00 = sm[((size_t)y0 * w + x0) * C + c];
    float v01 = sm[((size_t)y0 * w + x1) * C + c];
    float v10 = sm[((size_t)y1 * w + x0) * C + c];
    float v11 = sm[((size_t)y1 * w + x1) * C + c];
    float t0 = v00 * (1.f - fx) + v01 * fx;
    float t1 = v10 * (1.f - fx) + v11 * fx;
    big[i] += t0 * (1.f - fy) + t1 * fy;
}

// ---------------------------------------------------------------------------
// WMMA implicit-GEMM conv, templated on <KS, IC, OCT> so tap/K loops fully
// unroll (36/72 WMMAs straight-line for the 3x3 convs) and index math is
// shift/mask only.  One wave computes OCT 16(oc)x16(px) tiles, reusing each
// activation B-fragment OCT times (activations dominate traffic; weights are
// tiny and WGP$/L2 resident).
// X: f16 NHWC zero-bordered by pad=(KS==3); Wt: f16 [OC16][taps*IC].
// D layout: lane<16 -> M=r, lane>=16 -> M=8+r, N=lane&15 (ISA 7.12.2)
// A layout: lane m(<16): K 0..7,16..23 ; lane m+16: K 8..15,24..31
// B layout: lane n(<16): K 0..7,16..23 @ col n ; lane n+16: K 8..15,24..31 @ col n
// ---------------------------------------------------------------------------
template <int KS, int IC, int OCT>
__global__ __launch_bounds__(256) void k_conv_wmma(
    const _Float16* __restrict__ X, const _Float16* __restrict__ Wt,
    const float* __restrict__ bias,
    float* __restrict__ outF32, _Float16* __restrict__ outF16,
    int H, int W, int OC16, int leaky, int outPad,
    int ogShift, int txShift)
{
    constexpr int taps = KS * KS;
    constexpr int pad  = (KS == 3) ? 1 : 0;
    constexpr int Ktot = taps * IC;
    int wave = (blockIdx.x * blockDim.x + threadIdx.x) >> 5;
    int lane = threadIdx.x & 31;
    int tilesX   = W >> 4;
    int ocGroups = OC16 >> (OCT == 2 ? 5 : 4);
    int nTiles   = H * tilesX * ocGroups;
    if (wave >= nTiles) return;                 // whole-wave uniform exit: EXEC all-1
    int og  = wave & (ocGroups - 1);
    int r   = wave >> ogShift;
    int xt  = r & (tilesX - 1);
    int y   = r >> txShift;
    int x0  = xt << 4;
    int oc0 = og << (OCT == 2 ? 5 : 4);
    int Wp  = W + 2 * pad;
    int col = lane & 15;
    int hi  = lane >> 4;
    const _Float16* wrow0 = Wt + (size_t)(oc0 + col) * Ktot + hi * 8;
    const _Float16* wrow1 = wrow0 + (size_t)16 * Ktot;   // used only when OCT==2

    v8f acc0 = {};
    v8f acc1 = {};
    #pragma unroll
    for (int tap = 0; tap < taps; ++tap) {
        constexpr int three = 3;
        const int dy = (KS == 3) ? (tap / three - 1) : 0;
        const int dx = (KS == 3) ? (tap % three - 1) : 0;
        const _Float16* xb = X + (size_t)((y + pad + dy) * Wp + (x0 + pad + dx) + col) * IC + hi * 8;
        #pragma unroll
        for (int k0 = 0; k0 < IC; k0 += 32) {
            v8h b0 = *(const v8h*)(xb + k0);
            v8h b1 = *(const v8h*)(xb + k0 + 16);
            v16h Bf;
            #pragma unroll
            for (int q = 0; q < 8; ++q) { Bf[q] = b0[q]; Bf[q + 8] = b1[q]; }
            {
                v8h a0 = *(const v8h*)(wrow0 + tap * IC + k0);
                v8h a1 = *(const v8h*)(wrow0 + tap * IC + k0 + 16);
                v16h Af;
                #pragma unroll
                for (int q = 0; q < 8; ++q) { Af[q] = a0[q]; Af[q + 8] = a1[q]; }
                acc0 = __builtin_amdgcn_wmma_f32_16x16x32_f16(
                           false, Af, false, Bf, (short)0, acc0, false, false);
            }
            if (OCT == 2) {
                v8h a0 = *(const v8h*)(wrow1 + tap * IC + k0);
                v8h a1 = *(const v8h*)(wrow1 + tap * IC + k0 + 16);
                v16h Af;
                #pragma unroll
                for (int q = 0; q < 8; ++q) { Af[q] = a0[q]; Af[q + 8] = a1[q]; }
                acc1 = __builtin_amdgcn_wmma_f32_16x16x32_f16(
                           false, Af, false, Bf, (short)0, acc1, false, false);
            }
        }
    }

    int px = x0 + col;
    #pragma unroll
    for (int t = 0; t < OCT; ++t) {
        v8f acc = (t == 0) ? acc0 : acc1;
        int ocw = oc0 + t * 16 + hi * 8;        // 8 contiguous channels per lane
        float res[8];
        #pragma unroll
        for (int q = 0; q < 8; ++q) {
            float v = acc[q] + bias[ocw + q];
            if (leaky) v = (v >= 0.f) ? v : 0.1f * v;
            res[q] = v;
        }
        if (outF32) {
            float* o = outF32 + (size_t)(y * W + px) * OC16 + ocw;
            #pragma unroll
            for (int q = 0; q < 8; ++q) o[q] = res[q];
        }
        if (outF16) {
            int Wo = W + 2 * outPad;
            _Float16* o = outF16 + (size_t)((y + outPad) * Wo + (px + outPad)) * OC16 + ocw;
            v8h hv;
            #pragma unroll
            for (int q = 0; q < 8; ++q) hv[q] = (_Float16)res[q];
            *(v8h*)o = hv;                      // single b128 store
        }
    }
}

template <int KS, int IC, int OCT>
static void launch_conv(const _Float16* X, const _Float16* Wt, const float* bias,
                        float* oF32, _Float16* oF16, int H, int W, int OC16,
                        int leaky, int outPad, hipStream_t stream) {
    int tilesX = W / 16;
    int ocG    = OC16 / (16 * OCT);
    int waves  = H * tilesX * ocG;
    int txShift = __builtin_ctz((unsigned)tilesX);
    int ogShift = __builtin_ctz((unsigned)ocG);
    int grid = (waves * 32 + 255) / 256;
    k_conv_wmma<KS, IC, OCT><<<grid, 256, 0, stream>>>(
        X, Wt, bias, oF32, oF16, H, W, OC16, leaky, outPad, ogShift, txShift);
}

// ---------------------------------------------------------------------------
// Decode + score init (also clears keep region of d_out)
// ---------------------------------------------------------------------------
__global__ void k_decode(const float* __restrict__ conf, const float* __restrict__ txty,
                         float* __restrict__ dout, float* __restrict__ skey,
                         int* __restrict__ sidx) {
    int i = blockIdx.x * blockDim.x + threadIdx.x;
    if (i >= 65536) return;
    if (i >= 64512) { skey[i] = -2.f; sidx[i] = i; return; }
    dout[322560 + i] = 0.f;                     // keep mask cleared
    int loc = i / 3, a = i % 3;
    int base, hs; float stride; int s;
    if (loc < 16384)      { s = 0; base = 0;     hs = 128; stride = 8.f;  }
    else if (loc < 20480) { s = 1; base = 16384; hs = 64;  stride = 16.f; }
    else                  { s = 2; base = 20480; hs = 32;  stride = 32.f; }
    int lp = loc - base;
    int gx = lp % hs, gy = lp / hs;
    const float A[9][2] = {{10,13},{16,30},{33,23},{30,61},{62,45},{59,119},
                           {116,90},{156,198},{373,326}};
    float aw = A[s * 3 + a][0], ah = A[s * 3 + a][1];
    float cf = conf[(size_t)loc * 16 + a];
    const float* t = txty + (size_t)loc * 16 + a * 4;
    float sx = 1.f / (1.f + expf(-t[0]));
    float sy = 1.f / (1.f + expf(-t[1]));
    float cx = (sx + gx) * stride;
    float cy = (sy + gy) * stride;
    float bw = expf(t[2]) * aw;
    float bh = expf(t[3]) * ah;
    const float inv = 1.f / 1024.f;
    float x1 = fminf(fmaxf((cx - 0.5f * bw) * inv, 0.f), 1.f);
    float y1 = fminf(fmaxf((cy - 0.5f * bh) * inv, 0.f), 1.f);
    float x2 = fminf(fmaxf((cx + 0.5f * bw) * inv, 0.f), 1.f);
    float y2 = fminf(fmaxf((cy + 0.5f * bh) * inv, 0.f), 1.f);
    dout[(size_t)i * 4 + 0] = x1;
    dout[(size_t)i * 4 + 1] = y1;
    dout[(size_t)i * 4 + 2] = x2;
    dout[(size_t)i * 4 + 3] = y2;
    float obj = 1.f / (1.f + expf(-cf));
    dout[258048 + i] = obj;
    skey[i] = (obj >= 0.01f) ? obj : -1.f;
    sidx[i] = i;
}

// descending bitonic sort step on 65536 (key,idx) pairs
__global__ void k_bitonic(float* __restrict__ key, int* __restrict__ idx, int j, int k) {
    int i = blockIdx.x * blockDim.x + threadIdx.x;
    int ixj = i ^ j;
    if (ixj <= i) return;
    bool descBlk = ((i & k) == 0);
    float a = key[i], b = key[ixj];
    bool swp = descBlk ? (a < b) : (a > b);
    if (swp) {
        key[i] = b; key[ixj] = a;
        int ta = idx[i]; idx[i] = idx[ixj]; idx[ixj] = ta;
    }
}

__global__ void k_topk_prep(const float* __restrict__ skey, const int* __restrict__ sidx,
                            const float* __restrict__ dout, float* __restrict__ bx,
                            float* __restrict__ area, int* __restrict__ cand) {
    int r = blockIdx.x * blockDim.x + threadIdx.x;
    if (r >= 1024) return;
    int id = sidx[r];
    float x1 = dout[(size_t)id * 4 + 0];
    float y1 = dout[(size_t)id * 4 + 1];
    float x2 = dout[(size_t)id * 4 + 2];
    float y2 = dout[(size_t)id * 4 + 3];
    bx[r * 4 + 0] = x1; bx[r * 4 + 1] = y1; bx[r * 4 + 2] = x2; bx[r * 4 + 3] = y2;
    area[r] = (x2 - x1) * (y2 - y1);
    cand[r] = (skey[r] >= 0.01f) ? 1 : 0;
}

__global__ void k_iou(const float* __restrict__ bx, const float* __restrict__ area,
                      float* __restrict__ iou) {
    int t = blockIdx.x * blockDim.x + threadIdx.x;
    if (t >= 1024 * 1024) return;
    int r = t >> 10, c = t & 1023;
    float iw = fmaxf(1e-28f, fminf(bx[r * 4 + 2], bx[c * 4 + 2]) - fmaxf(bx[r * 4 + 0], bx[c * 4 + 0]));
    float ih = fmaxf(1e-28f, fminf(bx[r * 4 + 3], bx[c * 4 + 3]) - fmaxf(bx[r * 4 + 1], bx[c * 4 + 1]));
    float inter = iw * ih;
    iou[t] = inter / (area[r] + area[c] - inter);
}

// greedy sequential NMS, single block of 1024 threads; scatters keep into d_out
__global__ __launch_bounds__(1024) void k_nms(const float* __restrict__ iou,
                                              const int* __restrict__ cand,
                                              const int* __restrict__ sidx,
                                              float* __restrict__ dout) {
    __shared__ int keep[1024];
    __shared__ int sup;
    int t = threadIdx.x;
    keep[t] = 0;
    __syncthreads();
    for (int i = 0; i < 1024; ++i) {
        if (t == 0) sup = 0;
        __syncthreads();
        if (keep[t] && iou[(size_t)t * 1024 + i] > 0.3f) sup = 1;
        __syncthreads();
        if (t == 0) keep[i] = (cand[i] && !sup) ? 1 : 0;
        __syncthreads();
    }
    if (keep[t]) dout[322560 + sidx[t]] = 1.0f;
}

// ---------------------------------------------------------------------------
// Host orchestration
// ---------------------------------------------------------------------------
extern "C" void kernel_launch(void* const* d_in, const int* in_sizes, int n_in,
                              void* d_out, int out_size, void* d_ws, size_t ws_size,
                              hipStream_t stream) {
    (void)in_sizes; (void)n_in; (void)out_size; (void)ws_size;
    const float* c3    = (const float*)d_in[0];
    const float* c4    = (const float*)d_in[1];
    const float* c5    = (const float*)d_in[2];
    const float* w_top = (const float*)d_in[3];  const float* b_top = (const float*)d_in[4];
    const float* w_lt4 = (const float*)d_in[5];  const float* b_lt4 = (const float*)d_in[6];
    const float* w_lt3 = (const float*)d_in[7];  const float* b_lt3 = (const float*)d_in[8];
    const float* w_sm3 = (const float*)d_in[9];  const float* b_sm3 = (const float*)d_in[10];
    const float* w_sm4 = (const float*)d_in[11]; const float* b_sm4 = (const float*)d_in[12];
    const float* w_sm5 = (const float*)d_in[13]; const float* b_sm5 = (const float*)d_in[14];
    const float* w_cf1 = (const float*)d_in[15]; const float* b_cf1 = (const float*)d_in[16];
    const float* w_cf2 = (const float*)d_in[17]; const float* b_cf2 = (const float*)d_in[18];
    const float* w_bb1 = (const float*)d_in[19]; const float* b_bb1 = (const float*)d_in[20];
    const float* w_bb2 = (const float*)d_in[21]; const float* b_bb2 = (const float*)d_in[22];
    float* dout = (float*)d_out;

    char* wsp = (char*)d_ws;
    size_t off = 0;
    auto alloc = [&](size_t bytes) -> void* {
        void* p = wsp + off;
        off = (off + bytes + 255) & ~(size_t)255;
        return p;
    };

    _Float16* c3h = (_Float16*)alloc((size_t)128 * 128 * 128 * 2);
    _Float16* c4h = (_Float16*)alloc((size_t)64 * 64 * 256 * 2);
    _Float16* c5h = (_Float16*)alloc((size_t)32 * 32 * 1024 * 2);

    _Float16* wtop16 = (_Float16*)alloc((size_t)128 * 1024 * 2);
    _Float16* wlt4_16 = (_Float16*)alloc((size_t)128 * 256 * 2);
    _Float16* wlt3_16 = (_Float16*)alloc((size_t)128 * 128 * 2);
    _Float16* wsm3_16 = (_Float16*)alloc((size_t)128 * 1152 * 2);
    _Float16* wsm4_16 = (_Float16*)alloc((size_t)128 * 1152 * 2);
    _Float16* wsm5_16 = (_Float16*)alloc((size_t)128 * 1152 * 2);
    _Float16* wcf1_16 = (_Float16*)alloc((size_t)128 * 1152 * 2);
    _Float16* wbb1_16 = (_Float16*)alloc((size_t)128 * 1152 * 2);
    _Float16* wcf2_16 = (_Float16*)alloc((size_t)16 * 128 * 2);
    _Float16* wbb2_16 = (_Float16*)alloc((size_t)16 * 128 * 2);

    float* btopP = (float*)alloc(128 * 4);
    float* blt4P = (float*)alloc(128 * 4);
    float* blt3P = (float*)alloc(128 * 4);
    float* bsm3P = (float*)alloc(128 * 4);
    float* bsm4P = (float*)alloc(128 * 4);
    float* bsm5P = (float*)alloc(128 * 4);
    float* bcf1P = (float*)alloc(128 * 4);
    float* bbb1P = (float*)alloc(128 * 4);
    float* bcf2P = (float*)alloc(128 * 4);
    float* bbb2P = (float*)alloc(128 * 4);

    float* p5f  = (float*)alloc((size_t)32 * 32 * 128 * 4);
    float* p4f  = (float*)alloc((size_t)64 * 64 * 128 * 4);       // lat4, then p4 in-place
    float* p3f  = (float*)alloc((size_t)128 * 128 * 128 * 4);     // lat3, then p3 in-place

    size_t padStart = off;
    _Float16* p3pad = (_Float16*)alloc((size_t)130 * 130 * 128 * 2);
    _Float16* p4pad = (_Float16*)alloc((size_t)66 * 66 * 128 * 2);
    _Float16* p5pad = (_Float16*)alloc((size_t)34 * 34 * 128 * 2);
    _Float16* s3pad = (_Float16*)alloc((size_t)130 * 130 * 128 * 2);
    _Float16* s4pad = (_Float16*)alloc((size_t)66 * 66 * 128 * 2);
    _Float16* s5pad = (_Float16*)alloc((size_t)34 * 34 * 128 * 2);
    size_t padEnd = off;

    _Float16* hcf = (_Float16*)alloc((size_t)128 * 128 * 128 * 2);
    _Float16* hbb = (_Float16*)alloc((size_t)128 * 128 * 128 * 2);
    float* confB = (float*)alloc((size_t)21504 * 16 * 4);
    float* txtyB = (float*)alloc((size_t)21504 * 16 * 4);
    float* skey  = (float*)alloc(65536 * 4);
    int*   sidx  = (int*)alloc(65536 * 4);
    float* bxT   = (float*)alloc(1024 * 4 * 4);
    float* areaT = (float*)alloc(1024 * 4);
    int*   candT = (int*)alloc(1024 * 4);
    float* iouM  = (float*)alloc((size_t)1024 * 1024 * 4);

    const int TB = 256;
    auto blks = [&](size_t n) { return (int)((n + TB - 1) / TB); };

    // 1) input conversions (batch 0 only — reference outputs use [0] only)
    k_nchw_to_nhwc_f16<<<blks((size_t)128 * 128 * 128), TB, 0, stream>>>(c3, c3h, 128, 128, 128);
    k_nchw_to_nhwc_f16<<<blks((size_t)256 * 64 * 64), TB, 0, stream>>>(c4, c4h, 256, 64, 64);
    k_nchw_to_nhwc_f16<<<blks((size_t)1024 * 32 * 32), TB, 0, stream>>>(c5, c5h, 1024, 32, 32);

    // 2) weight conversions
    k_cvt_w<<<blks((size_t)128 * 1024), TB, 0, stream>>>(w_top, b_top, wtop16, btopP, 128, 128, 1024, 1);
    k_cvt_w<<<blks((size_t)128 * 256), TB, 0, stream>>>(w_lt4, b_lt4, wlt4_16, blt4P, 128, 128, 256, 1);
    k_cvt_w<<<blks((size_t)128 * 128), TB, 0, stream>>>(w_lt3, b_lt3, wlt3_16, blt3P, 128, 128, 128, 1);
    k_cvt_w<<<blks((size_t)128 * 1152), TB, 0, stream>>>(w_sm3, b_sm3, wsm3_16, bsm3P, 128, 128, 128, 3);
    k_cvt_w<<<blks((size_t)128 * 1152), TB, 0, stream>>>(w_sm4, b_sm4, wsm4_16, bsm4P, 128, 128, 128, 3);
    k_cvt_w<<<blks((size_t)128 * 1152), TB, 0, stream>>>(w_sm5, b_sm5, wsm5_16, bsm5P, 128, 128, 128, 3);
    k_cvt_w<<<blks((size_t)128 * 1152), TB, 0, stream>>>(w_cf1, b_cf1, wcf1_16, bcf1P, 128, 128, 128, 3);
    k_cvt_w<<<blks((size_t)128 * 1152), TB, 0, stream>>>(w_bb1, b_bb1, wbb1_16, bbb1P, 128, 128, 128, 3);
    k_cvt_w<<<blks((size_t)16 * 128), TB, 0, stream>>>(w_cf2, b_cf2, wcf2_16, bcf2P, 3, 16, 128, 1);
    k_cvt_w<<<blks((size_t)16 * 128), TB, 0, stream>>>(w_bb2, b_bb2, wbb2_16, bbb2P, 12, 16, 128, 1);

    // 3) zero the padded-f16 region (borders stay zero -> branch-free 3x3 reads)
    size_t n32 = (padEnd - padStart) / 4;
    k_zero32<<<2048, TB, 0, stream>>>((uint32_t*)(wsp + padStart), n32);

    // 4) 1x1 lateral/top convs (WMMA GEMM, 2 oc-tiles/wave)
    launch_conv<1, 1024, 2>(c5h, wtop16, btopP, p5f, nullptr, 32, 32, 128, 0, 0, stream);
    launch_conv<1, 256, 2>(c4h, wlt4_16, blt4P, p4f, nullptr, 64, 64, 128, 0, 0, stream);
    launch_conv<1, 128, 2>(c3h, wlt3_16, blt3P, p3f, nullptr, 128, 128, 128, 0, 0, stream);

    // 5) FPN upsample-adds (in-place on the lateral buffers)
    k_upsample_add<<<blks((size_t)64 * 64 * 128), TB, 0, stream>>>(p5f, p4f, 128, 32, 32, 64, 64);
    k_upsample_add<<<blks((size_t)128 * 128 * 128), TB, 0, stream>>>(p4f, p3f, 128, 64, 64, 128, 128);

    // 6) pack to padded f16 for the 3x3 convs
    k_pack_f16pad<<<blks((size_t)32 * 32 * 128), TB, 0, stream>>>(p5f, p5pad, 128, 32, 32);
    k_pack_f16pad<<<blks((size_t)64 * 64 * 128), TB, 0, stream>>>(p4f, p4pad, 128, 64, 64);
    k_pack_f16pad<<<blks((size_t)128 * 128 * 128), TB, 0, stream>>>(p3f, p3pad, 128, 128, 128);

    // 7) 3x3 smoothing convs -> padded f16 (head inputs); 72 WMMAs/wave straight-line
    launch_conv<3, 128, 2>(p3pad, wsm3_16, bsm3P, nullptr, s3pad, 128, 128, 128, 0, 1, stream);
    launch_conv<3, 128, 2>(p4pad, wsm4_16, bsm4P, nullptr, s4pad, 64, 64, 128, 0, 1, stream);
    launch_conv<3, 128, 2>(p5pad, wsm5_16, bsm5P, nullptr, s5pad, 32, 32, 128, 0, 1, stream);

    // 8) heads per scale: 3x3 leaky -> f16, then 1x1 -> f32 (OC padded to 16)
    struct Sc { _Float16* x; int hs; int base; };
    Sc sc[3] = { {s3pad, 128, 0}, {s4pad, 64, 16384}, {s5pad, 32, 20480} };
    for (int s = 0; s < 3; ++s) {
        int hs = sc[s].hs;
        launch_conv<3, 128, 2>(sc[s].x, wcf1_16, bcf1P, nullptr, hcf, hs, hs, 128, 1, 0, stream);
        launch_conv<1, 128, 1>(hcf, wcf2_16, bcf2P, confB + (size_t)sc[s].base * 16, nullptr,
                               hs, hs, 16, 0, 0, stream);
        launch_conv<3, 128, 2>(sc[s].x, wbb1_16, bbb1P, nullptr, hbb, hs, hs, 128, 1, 0, stream);
        launch_conv<1, 128, 1>(hbb, wbb2_16, bbb2P, txtyB + (size_t)sc[s].base * 16, nullptr,
                               hs, hs, 16, 0, 0, stream);
    }

    // 9) decode boxes/scores directly into d_out + init sort keys
    k_decode<<<65536 / TB, TB, 0, stream>>>(confB, txtyB, dout, skey, sidx);

    // 10) descending bitonic sort (top-1024 of 65536)
    for (int k = 2; k <= 65536; k <<= 1)
        for (int j = k >> 1; j > 0; j >>= 1)
            k_bitonic<<<65536 / TB, TB, 0, stream>>>(skey, sidx, j, k);

    // 11) NMS
    k_topk_prep<<<1024 / TB, TB, 0, stream>>>(skey, sidx, dout, bxT, areaT, candT);
    k_iou<<<(1024 * 1024) / TB, TB, 0, stream>>>(bxT, areaT, iouM);
    k_nms<<<1, 1024, 0, stream>>>(iouM, candT, sidx, dout);
}
